// NegativePartialLogLikelihoodForCoxPH_57921928954503
// MI455X (gfx1250) — matrix operand order repaired and checked by reading, not compiled, not run on hardware
//
#include <hip/hip_runtime.h>
#include <hip/hip_bf16.h>
#include <math.h>

typedef __attribute__((ext_vector_type(16))) _Float16 v16h;
typedef __attribute__((ext_vector_type(8)))  float    v8f;

#define TJ    2048          // j-tile staged in LDS per block iteration
#define WAVES 8             // waves per block (wave32) -> 256 threads
#define BLK   (WAVES * 32)

// ---------------------------------------------------------------------------
// Kernel 1: global max of risks -> ws[0]  (single block, deterministic)
// ---------------------------------------------------------------------------
__global__ void cox_max_kernel(const float* __restrict__ risks,
                               float* __restrict__ wsM, int n) {
  __shared__ float s[1024];
  float m = -3.402823466e38f;
  for (int i = threadIdx.x; i < n; i += blockDim.x) m = fmaxf(m, risks[i]);
  s[threadIdx.x] = m;
  __syncthreads();
  for (int off = 512; off > 0; off >>= 1) {
    if (threadIdx.x < off) s[threadIdx.x] = fmaxf(s[threadIdx.x], s[threadIdx.x + off]);
    __syncthreads();
  }
  if (threadIdx.x == 0) wsM[0] = s[0];
}

// ---------------------------------------------------------------------------
// Kernel 2: contiguous times copy + e[j] = exp(r_j - M) in f16
// ---------------------------------------------------------------------------
__global__ void cox_prep_kernel(const float* __restrict__ input,
                                const float* __restrict__ target,
                                const float* __restrict__ wsM,
                                float* __restrict__ times_c,
                                _Float16* __restrict__ e_h, int n) {
  int j = blockIdx.x * blockDim.x + threadIdx.x;
  if (j < n) {
    float M = wsM[0];
    times_c[j] = target[2 * j + 1];
    e_h[j] = (_Float16)expf(input[j] - M);
  }
}

// ---------------------------------------------------------------------------
// Kernel 3: masked GEMV via V_WMMA_F32_16X16X32_F16.
// Each wave owns one 16-row i-tile. A = 0/1 mask (f32 compares, exact ties),
// B = e[j] broadcast across columns. D column 0 = S[i].
// ---------------------------------------------------------------------------
__global__ __launch_bounds__(BLK)
void cox_wmma_kernel(const float* __restrict__ input,
                     const float* __restrict__ target,
                     const float* __restrict__ wsM,
                     const float* __restrict__ times_c,
                     const _Float16* __restrict__ e_h,
                     float* __restrict__ partials, int n) {
  __shared__ float    s_t[TJ];
  __shared__ _Float16 s_e[TJ];
  __shared__ float    s_red[WAVES * 2];

  const int tid  = threadIdx.x;
  const int lane = tid & 31;
  const int wave = tid >> 5;
  const int hi   = lane >> 4;                 // 0: lanes 0-15, 1: lanes 16-31
  const int i0   = (blockIdx.x * WAVES + wave) * 16;
  const int irow = i0 + (lane & 15);
  // rows past n get +inf time -> empty mask row (contributes nothing)
  const float my_ti = (irow < n) ? times_c[irow] : __int_as_float(0x7f800000);

  union { v8f v; float f[8]; } acc;
  acc.v = (v8f){0.f, 0.f, 0.f, 0.f, 0.f, 0.f, 0.f, 0.f};

  for (int j0 = 0; j0 < n; j0 += TJ) {
    __syncthreads();
    for (int k = tid; k < TJ; k += BLK) {     // uniform trip count (TJ % BLK == 0)
      int j = j0 + k;
      bool ok = (j < n);
      s_t[k] = ok ? times_c[j] : 0.0f;
      s_e[k] = ok ? e_h[j] : (_Float16)0.0f;  // e=0 pad -> zero contribution
    }
    __syncthreads();

#pragma unroll 4
    for (int jb = 0; jb < TJ; jb += 32) {
      union { v16h v; _Float16 h[16]; } a, b;
#pragma unroll
      for (int h = 0; h < 16; ++h) {
        // A layout (16x32 f16): half h -> K = (h<8 ? h : h+8) + 8*hi
        int ka = (h < 8 ? h : 8 + h) + 8 * hi;
        a.h[h] = (s_t[jb + ka] >= my_ti) ? (_Float16)1.0f : (_Float16)0.0f;
        // B layout (32x16 f16): half h -> K = (h<8 ? h : h+8) + 8*hi, broadcast over columns
        b.h[h] = s_e[jb + (h < 8 ? h : 8 + h) + 8 * hi];
      }
      acc.v = __builtin_amdgcn_wmma_f32_16x16x32_f16(
          false, a.v, false, b.v, (short)0, acc.v, false, false);
    }
  }

  // Epilogue (after all WMMAs): column N=0 lives in lanes 0 (rows 0-7 in
  // v0..v7) and 16 (rows 8-15).
  if ((lane & 15) == 0) {
    const float M = wsM[0];
    float local = 0.0f;
#pragma unroll
    for (int r = 0; r < 8; ++r) {
      int i = i0 + hi * 8 + r;
      if (i < n) {
        float lse = logf(acc.f[r]) + M;       // S >= e[i] > 0 always
        float status = target[2 * i];
        if (status == 0.0f) local += (lse - input[i]);
      }
    }
    s_red[wave * 2 + hi] = local;
  }
  __syncthreads();
  if (tid == 0) {
    float s = 0.f;
    for (int k = 0; k < WAVES * 2; ++k) s += s_red[k];
    partials[blockIdx.x] = s;                 // deterministic slot, no atomics
  }
}

// ---------------------------------------------------------------------------
// Kernel 4: ordered (deterministic) sum of block partials -> mean
// ---------------------------------------------------------------------------
__global__ void cox_final_kernel(const float* __restrict__ partials,
                                 int nparts, float invN,
                                 float* __restrict__ out) {
  if (threadIdx.x == 0 && blockIdx.x == 0) {
    float s = 0.f;
    for (int i = 0; i < nparts; ++i) s += partials[i];
    out[0] = s * invN;
  }
}

// ---------------------------------------------------------------------------
extern "C" void kernel_launch(void* const* d_in, const int* in_sizes, int n_in,
                              void* d_out, int out_size, void* d_ws, size_t ws_size,
                              hipStream_t stream) {
  const float* input  = (const float*)d_in[0];   // (N,1) risks
  const float* target = (const float*)d_in[1];   // (N,2) [status, time]
  float* out = (float*)d_out;
  int n = in_sizes[0];

  // Workspace layout (floats): [0]=M, [16..16+2048)=block partials,
  // [2064..2064+n)=contiguous times, then n halves of e.
  float* ws_f     = (float*)d_ws;
  float* wM       = ws_f;
  float* partials = ws_f + 16;
  float* times_c  = ws_f + 16 + 2048;
  _Float16* e_h   = (_Float16*)(times_c + n);

  cox_max_kernel<<<1, 1024, 0, stream>>>(input, wM, n);

  int pb = (n + 255) / 256;
  cox_prep_kernel<<<pb, 256, 0, stream>>>(input, target, wM, times_c, e_h, n);

  int nb = (n + (16 * WAVES) - 1) / (16 * WAVES);
  cox_wmma_kernel<<<nb, BLK, 0, stream>>>(input, target, wM, times_c, e_h,
                                          partials, n);

  cox_final_kernel<<<1, 32, 0, stream>>>(partials, nb, 1.0f / (float)n, out);
}
